// APPNP_48868137893893
// MI455X (gfx1250) — matrix operand order repaired and checked by reading, not compiled, not run on hardware
//
#include <hip/hip_runtime.h>
#include <hip/hip_bf16.h>

// ---------------------------------------------------------------------------
// APPNP on prefix tree, MI455X (gfx1250, wave32, WMMA bf16)
//   d      = rsqrt(rowsum(adj))
//   a_bf16 = bf16(d_i * adj_ij * d_j)          (128 MB, L2-resident)
//   h0     = relu(X @ W + b)                   (bf16 WMMA, fp32 accum)
//   6x:  h = 0.8*(a_bf16 @ h) + 0.2*h0        (bf16 WMMA, fp32 accum)
// h kept transposed bf16 [256][8192] between layers. Propagation kernel
// pipelines: B tile staged in LDS via global_load_async_to_lds_b128
// (double-buffered, ASYNCcnt), A fragments register double-buffered.
// ---------------------------------------------------------------------------

typedef __bf16 v16bf __attribute__((ext_vector_type(16)));
typedef float  v8f   __attribute__((ext_vector_type(8)));
typedef unsigned int v4u __attribute__((ext_vector_type(4)));

#define N_NODES 8192
#define EMBDIM  1024
#define TREEHID 256
#define NLAYER  6
#define ALPHA   0.2f

#define LDS_ROW 80                 // 64 data bytes + 16 pad (bank de-conflict)
#define LDS_BUF (64 * LDS_ROW)     // one K-step B tile: 64 rows(n) x 32 k bf16

struct U8 { v4u lo; v4u hi; };           // 32 bytes == v16bf
struct W8 { unsigned int w[8]; };        // 32 bytes == v16bf

__device__ __forceinline__ unsigned short f32_to_bf16(float f) {
    unsigned int u = __builtin_bit_cast(unsigned int, f);
    unsigned int r = u + 0x7FFFu + ((u >> 16) & 1u);   // round-to-nearest-even
    return (unsigned short)(r >> 16);
}

__device__ __forceinline__ v8f wmma_bf16(v16bf a, v16bf b, v8f c) {
    // (neg_a, A, neg_b, B, c_mod, C, reuse_a, reuse_b)
    return __builtin_amdgcn_wmma_f32_16x16x32_bf16(false, a, false, b,
                                                   (short)0, c, false, false);
}

// Async global -> LDS copy, 16 bytes per lane. lds_off is a raw 32-bit LDS
// byte offset (low 32 bits of the generic shared address). ASYNCcnt-tracked.
__device__ __forceinline__ void async_b128(unsigned lds_off, const void* gptr) {
    asm volatile("global_load_async_to_lds_b128 %0, %1, off"
                 :: "v"(lds_off), "v"(gptr) : "memory");
}
__device__ __forceinline__ void wait_async0() {
    asm volatile("s_wait_asynccnt 0x0" ::: "memory");
}

// A-fragment 16x32 bf16 from row-major bf16 matrix, tile origin (row0, k0).
// Lane l<16: M=l, K = k0+0..15.
// Lane l>=16: M=l-16, K = k0+16..31.
__device__ __forceinline__ v16bf load_a_bf16(const unsigned short* __restrict__ A,
                                             long ld, int row0, int k0) {
    int lane = threadIdx.x & 31;
    int half = lane >> 4;
    long row = row0 + (lane & 15);
    const unsigned short* p = A + row * ld + k0 + half * 16;
    U8 u;
    u.lo = *(const v4u*)(p);
    u.hi = *(const v4u*)(p + 8);
    return __builtin_bit_cast(v16bf, u);
}

// A-fragment with on-the-fly fp32 -> bf16 conversion (for X @ W).
__device__ __forceinline__ v16bf load_a_f32(const float* __restrict__ A,
                                            long ld, int row0, int k0) {
    int lane = threadIdx.x & 31;
    int half = lane >> 4;
    long row = row0 + (lane & 15);
    const float* p = A + row * ld + k0 + half * 16;
    W8 u;
#pragma unroll
    for (int i = 0; i < 4; ++i)
        u.w[i] = (unsigned int)f32_to_bf16(p[2 * i]) |
                 ((unsigned int)f32_to_bf16(p[2 * i + 1]) << 16);
#pragma unroll
    for (int i = 0; i < 4; ++i)
        u.w[4 + i] = (unsigned int)f32_to_bf16(p[8 + 2 * i]) |
                     ((unsigned int)f32_to_bf16(p[8 + 2 * i + 1]) << 16);
    return __builtin_bit_cast(v16bf, u);
}

// B-fragment 32x16 bf16 from K-contiguous (transposed) global storage BT[n][k].
// Lane l<16: col n0+l, K = k0+0..15; lane l>=16: col n0+l-16, K = k0+16..31.
__device__ __forceinline__ v16bf load_b_bf16(const unsigned short* __restrict__ BT,
                                             long ld, int n0, int k0) {
    int lane = threadIdx.x & 31;
    int half = lane >> 4;
    long n = n0 + (lane & 15);
    const unsigned short* p = BT + n * ld + k0 + half * 16;
    U8 u;
    u.lo = *(const v4u*)(p);
    u.hi = *(const v4u*)(p + 8);
    return __builtin_bit_cast(v16bf, u);
}

// B-fragment from the LDS-staged tile (rows padded to LDS_ROW bytes).
__device__ __forceinline__ v16bf load_b_lds(const unsigned char* base, int j) {
    int lane = threadIdx.x & 31;
    int half = lane >> 4, nl = lane & 15;
    const unsigned char* p = base + (j * 16 + nl) * LDS_ROW + half * 32;
    U8 u;
    u.lo = *(const v4u*)(p);
    u.hi = *(const v4u*)(p + 16);
    return __builtin_bit_cast(v16bf, u);
}

// ---------------------------------------------------------------------------
// Kernel 1: d[row] = rsqrt(sum_j adj[row][j])
// ---------------------------------------------------------------------------
__global__ void k_degree(const float* __restrict__ adj, float* __restrict__ d) {
    __shared__ float red[256];
    long row = blockIdx.x;
    const float* p = adj + row * (long)N_NODES;
    float s = 0.f;
    for (int c = threadIdx.x; c < N_NODES; c += 256) s += p[c];
    red[threadIdx.x] = s;
    __syncthreads();
    for (int off = 128; off > 0; off >>= 1) {
        if (threadIdx.x < off) red[threadIdx.x] += red[threadIdx.x + off];
        __syncthreads();
    }
    if (threadIdx.x == 0) d[row] = rsqrtf(red[0]);
}

// ---------------------------------------------------------------------------
// Kernel 2: a_bf16[i][j] = bf16(d_i * adj[i][j] * d_j)   (8 elems / thread)
// ---------------------------------------------------------------------------
__global__ void k_scale_cast(const float* __restrict__ adj,
                             const float* __restrict__ d,
                             unsigned short* __restrict__ abf) {
    long row = blockIdx.y;
    int col0 = (blockIdx.x * 256 + threadIdx.x) * 8;
    float dr = d[row];
    const float4* pa = (const float4*)(adj + row * (long)N_NODES + col0);
    const float4* pd = (const float4*)(d + col0);
    float4 a0 = pa[0], a1 = pa[1];
    float4 d0 = pd[0], d1 = pd[1];
    v4u w;
    w.x = (unsigned int)f32_to_bf16(dr * a0.x * d0.x) |
          ((unsigned int)f32_to_bf16(dr * a0.y * d0.y) << 16);
    w.y = (unsigned int)f32_to_bf16(dr * a0.z * d0.z) |
          ((unsigned int)f32_to_bf16(dr * a0.w * d0.w) << 16);
    w.z = (unsigned int)f32_to_bf16(dr * a1.x * d1.x) |
          ((unsigned int)f32_to_bf16(dr * a1.y * d1.y) << 16);
    w.w = (unsigned int)f32_to_bf16(dr * a1.z * d1.z) |
          ((unsigned int)f32_to_bf16(dr * a1.w * d1.w) << 16);
    *(v4u*)(abf + row * (long)N_NODES + col0) = w;
}

// ---------------------------------------------------------------------------
// Kernel 3: WT[n][k] = bf16(W[k][n])   (transpose + cast, tiny)
// ---------------------------------------------------------------------------
__global__ void k_wt(const float* __restrict__ W, unsigned short* __restrict__ WT) {
    int idx = blockIdx.x * 256 + threadIdx.x;          // 256*1024 elements
    int n = idx / EMBDIM;
    int k = idx % EMBDIM;
    WT[(long)n * EMBDIM + k] = f32_to_bf16(W[(long)k * TREEHID + n]);
}

// ---------------------------------------------------------------------------
// Kernel 4: h0 = relu(X @ W + b); also emit hT0 = bf16(h0)^T
//   block = 256 threads = 8 waves; each wave owns a 32x64 output tile.
// ---------------------------------------------------------------------------
__global__ void k_gemm_h0(const float* __restrict__ X,
                          const unsigned short* __restrict__ WT,
                          const float* __restrict__ bias,
                          float* __restrict__ h0,
                          unsigned short* __restrict__ hT0) {
    int wave = threadIdx.x >> 5;
    int m0 = (blockIdx.x * 8 + wave) * 32;
    int n0 = blockIdx.y * 64;
    v8f acc[2][4] = {};

    for (int k0 = 0; k0 < EMBDIM; k0 += 32) {
        v16bf a0 = load_a_f32(X, EMBDIM, m0, k0);
        v16bf a1 = load_a_f32(X, EMBDIM, m0 + 16, k0);
#pragma unroll
        for (int j = 0; j < 4; ++j) {
            v16bf b = load_b_bf16(WT, EMBDIM, n0 + j * 16, k0);
            acc[0][j] = wmma_bf16(a0, b, acc[0][j]);
            acc[1][j] = wmma_bf16(a1, b, acc[1][j]);
        }
    }

    int lane = threadIdx.x & 31, half = lane >> 4, nl = lane & 15;
#pragma unroll
    for (int i = 0; i < 2; ++i)
#pragma unroll
        for (int j = 0; j < 4; ++j) {
            int n = n0 + j * 16 + nl;
            float bv = bias[n];
#pragma unroll
            for (int r = 0; r < 8; ++r) {
                int m = m0 + i * 16 + r + half * 8;
                float v = acc[i][j][r] + bv;
                v = fmaxf(v, 0.f);
                h0[(long)m * TREEHID + n] = v;
                hT0[(long)n * N_NODES + m] = f32_to_bf16(v);
            }
        }
}

// ---------------------------------------------------------------------------
// Kernel 5: one APPNP step: h = 0.8*(a_bf16 @ hTin^T) + 0.2*h0
//   block = 256 threads = 8 waves, one 32x64 output tile per wave, shared
//   64-col B strip. B double-buffered in LDS via async global->LDS copies;
//   A fragments register double-buffered. 8 v_wmma per 32-wide K step.
// ---------------------------------------------------------------------------
__global__ void k_prop(const unsigned short* __restrict__ abf,
                       const unsigned short* __restrict__ hTin,
                       const float* __restrict__ h0,
                       unsigned short* __restrict__ hTout,
                       float* __restrict__ out,
                       int last) {
    __shared__ __align__(16) unsigned char ldsB[2 * LDS_BUF];

    int wave = threadIdx.x >> 5;
    int m0 = (blockIdx.x * 8 + wave) * 32;
    int n0 = blockIdx.y * 64;
    int lane = threadIdx.x & 31, half = lane >> 4, nl = lane & 15;
    v8f acc[2][4] = {};

    // Per-thread async-copy assignment: 256 threads x 16B = 64 rows x 64B.
    int cr = threadIdx.x >> 2;            // row (n) within strip: 0..63
    int cs = threadIdx.x & 3;             // 16-byte segment: 0..3
    unsigned ldsbase = (unsigned)(unsigned long long)(const void*)&ldsB[0];
    unsigned mydst = ldsbase + cr * LDS_ROW + cs * 16;
    const unsigned short* mysrc =
        hTin + (long)(n0 + cr) * N_NODES + cs * 8;

    // Prologue: stage k0 = 0 into buffer 0, preload first A fragments.
    async_b128(mydst, mysrc);
    v16bf a0 = load_a_bf16(abf, N_NODES, m0, 0);
    v16bf a1 = load_a_bf16(abf, N_NODES, m0 + 16, 0);
    wait_async0();
    __syncthreads();

    for (int k0 = 0; k0 < N_NODES; k0 += 32) {
        int buf = (k0 >> 5) & 1;
        const unsigned char* cur = ldsB + buf * LDS_BUF;

        // Stage next K-step (overshoot on last iter stays inside workspace).
        async_b128(mydst + (buf ^ 1) * LDS_BUF, mysrc + k0 + 32);
        __builtin_prefetch(abf + (long)(m0 + nl) * N_NODES + k0 + 1024, 0, 1);
        v16bf na0 = load_a_bf16(abf, N_NODES, m0, k0 + 32);
        v16bf na1 = load_a_bf16(abf, N_NODES, m0 + 16, k0 + 32);

#pragma unroll
        for (int j = 0; j < 4; ++j) {
            v16bf b = load_b_lds(cur, j);
            acc[0][j] = wmma_bf16(a0, b, acc[0][j]);
            acc[1][j] = wmma_bf16(a1, b, acc[1][j]);
        }

        wait_async0();        // next buffer fully staged
        __syncthreads();      // publish LDS writes / retire LDS reads
        a0 = na0;
        a1 = na1;
    }

#pragma unroll
    for (int i = 0; i < 2; ++i)
#pragma unroll
        for (int j = 0; j < 4; ++j) {
            int n = n0 + j * 16 + nl;
#pragma unroll
            for (int r = 0; r < 8; ++r) {
                int m = m0 + i * 16 + r + half * 8;
                float v = (1.0f - ALPHA) * acc[i][j][r] +
                          ALPHA * h0[(long)m * TREEHID + n];
                if (last) out[(long)m * TREEHID + n] = v;
                else      hTout[(long)n * N_NODES + m] = f32_to_bf16(v);
            }
        }
}

// ---------------------------------------------------------------------------
// Host-side orchestration (graph-capture safe: only async launches on stream)
// ---------------------------------------------------------------------------
extern "C" void kernel_launch(void* const* d_in, const int* in_sizes, int n_in,
                              void* d_out, int out_size, void* d_ws, size_t ws_size,
                              hipStream_t stream) {
    const float* X    = (const float*)d_in[0];   // [8192,1024]
    const float* W    = (const float*)d_in[1];   // [1024,256]
    const float* bias = (const float*)d_in[2];   // [256]
    const float* adj  = (const float*)d_in[3];   // [8192,8192]
    float* out        = (float*)d_out;           // [8192,256]

    char* ws = (char*)d_ws;
    size_t off = 0;
    unsigned short* abf = (unsigned short*)(ws + off); off += (size_t)N_NODES * N_NODES * 2; // 128 MB
    unsigned short* hT0 = (unsigned short*)(ws + off); off += (size_t)TREEHID * N_NODES * 2; // 4 MB
    unsigned short* hT1 = (unsigned short*)(ws + off); off += (size_t)TREEHID * N_NODES * 2; // 4 MB
    float* h0           = (float*)(ws + off);          off += (size_t)N_NODES * TREEHID * 4; // 8 MB
    unsigned short* WT  = (unsigned short*)(ws + off); off += (size_t)TREEHID * EMBDIM * 2;  // 0.5 MB
    float* dvec         = (float*)(ws + off);          off += (size_t)N_NODES * 4;

    k_degree    <<<N_NODES, 256, 0, stream>>>(adj, dvec);
    k_scale_cast<<<dim3(4, N_NODES), 256, 0, stream>>>(adj, dvec, abf);
    k_wt        <<<(TREEHID * EMBDIM) / 256, 256, 0, stream>>>(W, WT);
    k_gemm_h0   <<<dim3(N_NODES / 256, TREEHID / 64), 256, 0, stream>>>(X, WT, bias, h0, hT0);

    unsigned short* hin = hT0;
    unsigned short* hout = hT1;
    for (int l = 0; l < NLAYER; ++l) {
        int last = (l == NLAYER - 1) ? 1 : 0;
        k_prop<<<dim3(N_NODES / 256, TREEHID / 64), 256, 0, stream>>>(
            abf, hin, h0, hout, out, last);
        unsigned short* t = hin; hin = hout; hout = t;
    }
    (void)in_sizes; (void)n_in; (void)out_size; (void)ws_size;
}
